// TransformerEncoder_57647051047153
// MI455X (gfx1250) — matrix operand order repaired and checked by reading, not compiled, not run on hardware
//
#include <hip/hip_runtime.h>

// Transformer encoder for MI455X (gfx1250), wave32 + WMMA bf16.
// GEMM-shaped math on v_wmma_f32_16x16x32_bf16; fragment gathers are 16B
// ds_load_b128/global_load_b128; GEMM tiles staged with async-to-LDS copies
// (ASYNCcnt) when the toolchain exposes them, double-buffered.

#define D_MODEL 1024
#define D_FF    4096
#define NHEADS  16
#define HDIM    64
#define NLAYERS 4
#define LN_EPS  1e-5f

typedef __attribute__((ext_vector_type(16))) __bf16        v16bf;
typedef __attribute__((ext_vector_type(8)))  float         v8f;
typedef __attribute__((ext_vector_type(4)))  unsigned int  u32x4;  // 16B POD vector
typedef __attribute__((ext_vector_type(4)))  int           i32x4;  // builtin arg type
typedef __attribute__((ext_vector_type(4)))  unsigned short u16x4;

union FragBF {
  v16bf v;
  u32x4 q[2];
  unsigned short s[16];
};

__device__ __forceinline__ unsigned short f32_bf16(float f) {
  unsigned int u = __float_as_uint(f);
  u += 0x7FFFu + ((u >> 16) & 1u);     // round-to-nearest-even
  return (unsigned short)(u >> 16);
}

__device__ __forceinline__ v8f wmma_bf16(const FragBF& a, const FragBF& b, v8f c) {
  return __builtin_amdgcn_wmma_f32_16x16x32_bf16(false, a.v, false, b.v,
                                                 (short)0, c, false, false);
}

// --- CDNA5 async global->LDS (GLOBAL_LOAD_ASYNC_TO_LDS_B128, ASYNCcnt) ------
#if defined(__has_builtin)
#if __has_builtin(__builtin_amdgcn_global_load_async_to_lds_b128) && \
    __has_builtin(__builtin_amdgcn_s_wait_asynccnt)
#define USE_ASYNC_LDS 1
#endif
#endif
#ifndef USE_ASYNC_LDS
#define USE_ASYNC_LDS 0
#endif

__device__ __forceinline__ void copy16_g2l(const u32x4* g, u32x4* l) {
#if USE_ASYNC_LDS
  // builtin expects non-const int4 pointers: AS(1) source, AS(3) destination
  __builtin_amdgcn_global_load_async_to_lds_b128(
      (__attribute__((address_space(1))) i32x4*)g,
      (__attribute__((address_space(3))) i32x4*)l, 0, 0);
#else
  *l = *g;
#endif
}

__device__ __forceinline__ void wait_async_copies() {
#if USE_ASYNC_LDS
  __builtin_amdgcn_s_wait_asynccnt(0);
#endif
}

// ---------------------------------------------------------------------------
// Fused fp32 -> bf16 + transpose (per z-slice): dst[z][c][r] = bf16(src[z][r][c])
// Produces K-major weights so GEMM B tiles are straight row-major copies.
// ---------------------------------------------------------------------------
__global__ __launch_bounds__(256)
void cvt_transpose_bf16_kernel(const float* __restrict__ src,
                               unsigned short* __restrict__ dst, int rows, int cols) {
  __shared__ unsigned short tile[32][33];
  const size_t zs = (size_t)blockIdx.z * rows * cols;
  const int c0 = blockIdx.x * 32, r0 = blockIdx.y * 32;
  const int tx = threadIdx.x, ty = threadIdx.y;
  #pragma unroll
  for (int i = 0; i < 32; i += 8)
    tile[ty + i][tx] = f32_bf16(src[zs + (size_t)(r0 + ty + i) * cols + c0 + tx]);
  __syncthreads();
  #pragma unroll
  for (int i = 0; i < 32; i += 8)
    dst[zs + (size_t)(c0 + ty + i) * rows + r0 + tx] = tile[tx][ty + i];
}

// ---------------------------------------------------------------------------
// bf16 transpose per batch: dst[(b*cols + c)*rows + r] = src[(b*rows+r)*cols+c]
// Gets V into hd-major layout for kv-contiguous WMMA B fragments.
// ---------------------------------------------------------------------------
__global__ __launch_bounds__(256)
void transpose_bf16_kernel(const unsigned short* __restrict__ src,
                           unsigned short* __restrict__ dst, int rows, int cols) {
  __shared__ unsigned short tile[32][33];
  const size_t base = (size_t)blockIdx.z * rows * cols;
  const int c0 = blockIdx.x * 32, r0 = blockIdx.y * 32;
  const int tx = threadIdx.x, ty = threadIdx.y;
  #pragma unroll
  for (int i = 0; i < 32; i += 8)
    tile[ty + i][tx] = src[base + (size_t)(r0 + ty + i) * cols + c0 + tx];
  __syncthreads();
  #pragma unroll
  for (int i = 0; i < 32; i += 8)
    dst[base + (size_t)(c0 + ty + i) * rows + r0 + tx] = tile[tx][ty + i];
}

// ---------------------------------------------------------------------------
// LayerNorm over D=1024, one row per block (256 threads, 8 wave32s).
// ---------------------------------------------------------------------------
__global__ __launch_bounds__(256)
void layernorm_bf16_kernel(const float* __restrict__ x, const float* __restrict__ g,
                           const float* __restrict__ b, unsigned short* __restrict__ out) {
  const int row  = blockIdx.x;
  const int t    = threadIdx.x;
  const int wave = t >> 5, lane = t & 31;
  const float4* x4 = (const float4*)(x + (size_t)row * D_MODEL);
  float4 v = x4[t];

  __shared__ float red[8];

  float s = v.x + v.y + v.z + v.w;
  #pragma unroll
  for (int o = 16; o > 0; o >>= 1) s += __shfl_xor(s, o, 32);
  if (lane == 0) red[wave] = s;
  __syncthreads();
  if (t == 0) { float a = 0.f; for (int i = 0; i < 8; ++i) a += red[i]; red[0] = a; }
  __syncthreads();
  const float mu = red[0] * (1.0f / D_MODEL);
  __syncthreads();

  float dx = v.x - mu, dy = v.y - mu, dz = v.z - mu, dw = v.w - mu;
  float q = dx * dx + dy * dy + dz * dz + dw * dw;
  #pragma unroll
  for (int o = 16; o > 0; o >>= 1) q += __shfl_xor(q, o, 32);
  if (lane == 0) red[wave] = q;
  __syncthreads();
  if (t == 0) { float a = 0.f; for (int i = 0; i < 8; ++i) a += red[i]; red[0] = a; }
  __syncthreads();
  const float rs = rsqrtf(red[0] * (1.0f / D_MODEL) + LN_EPS);

  const int c = t * 4;
  u16x4 o4;
  o4[0] = f32_bf16(dx * rs * g[c + 0] + b[c + 0]);
  o4[1] = f32_bf16(dy * rs * g[c + 1] + b[c + 1]);
  o4[2] = f32_bf16(dz * rs * g[c + 2] + b[c + 2]);
  o4[3] = f32_bf16(dw * rs * g[c + 3] + b[c + 3]);
  *(u16x4*)(out + (size_t)row * D_MODEL + c) = o4;
}

// ---------------------------------------------------------------------------
// WMMA GEMM: C[M,N] = A_bf16[M,K] @ WT_bf16[N][K]^T + bias (+ReLU) (+resid).
// 128x128 block tile, K stepped by 64, double-buffered async-to-LDS staging.
// 8 waves, each owns 32x64: 2 A frags x 4 B frags x 2 k-chunks = 16 WMMAs
// between barriers. Both tiles are row-major 16B copies (WT is K-major).
// ---------------------------------------------------------------------------
template <bool RELU, bool HASRES, bool WRITEF, bool WRITEBF>
__global__ __launch_bounds__(256)
void gemm_bf16_kernel(const unsigned short* __restrict__ A,
                      const unsigned short* __restrict__ WT,
                      const float* __restrict__ bias,
                      const float* __restrict__ resid,
                      float* __restrict__ outf,
                      unsigned short* __restrict__ outbf,
                      int N, int K) {
  constexpr int BM = 128, BN = 128, BK = 64, LDK = BK + 8;  // 144B row stride
  __shared__ unsigned short As[2][BM][LDK];
  __shared__ unsigned short Bs[2][BN][LDK];

  const int tid  = threadIdx.x;
  const int wave = tid >> 5, lane = tid & 31;
  const int lhi  = lane >> 4, llo = lane & 15;
  const int m0   = blockIdx.y * BM, n0 = blockIdx.x * BN;
  const int wm   = (wave >> 1) * 32;   // 4 wave rows of 32
  const int wn   = (wave & 1) * 64;    // 2 wave cols of 64

  v8f acc[2][4];
  #pragma unroll
  for (int mi = 0; mi < 2; ++mi)
    #pragma unroll
    for (int nb = 0; nb < 4; ++nb) acc[mi][nb] = (v8f){};

  const u32x4* Ag = (const u32x4*)A;
  const u32x4* Wg = (const u32x4*)WT;

  auto stage = [&](int buf, int k0) {
    for (int i = tid; i < BM * (BK / 8); i += 256) {   // A: 128 rows x 8 chunks
      int r = i >> 3, qc = i & 7;
      copy16_g2l(Ag + ((size_t)(m0 + r) * K + k0) / 8 + qc,
                 (u32x4*)&As[buf][r][qc * 8]);
    }
    for (int i = tid; i < BN * (BK / 8); i += 256) {   // B: 128 N-rows x 8 chunks
      int r = i >> 3, qc = i & 7;
      copy16_g2l(Wg + ((size_t)(n0 + r) * K + k0) / 8 + qc,
                 (u32x4*)&Bs[buf][r][qc * 8]);
    }
  };

  const int nsteps = K / BK;
  stage(0, 0);
  wait_async_copies();
  __syncthreads();

  for (int s = 0; s < nsteps; ++s) {
    const int buf = s & 1;
    if (s + 1 < nsteps) stage(buf ^ 1, (s + 1) * BK);  // prefetch next tiles

    #pragma unroll
    for (int kk = 0; kk < 2; ++kk) {
      FragBF a0, a1, bf[4];
      const u32x4* ap0 = (const u32x4*)&As[buf][wm + llo][0];
      const u32x4* ap1 = (const u32x4*)&As[buf][wm + 16 + llo][0];
      a0.q[0] = ap0[kk * 4 + 2 * lhi];  a0.q[1] = ap0[kk * 4 + 2 * lhi + 1];
      a1.q[0] = ap1[kk * 4 + 2 * lhi];  a1.q[1] = ap1[kk * 4 + 2 * lhi + 1];
      #pragma unroll
      for (int nb = 0; nb < 4; ++nb) {
        const u32x4* bp = (const u32x4*)&Bs[buf][wn + nb * 16 + llo][0];
        bf[nb].q[0] = bp[kk * 4 + 2 * lhi];
        bf[nb].q[1] = bp[kk * 4 + 2 * lhi + 1];
      }
      #pragma unroll
      for (int nb = 0; nb < 4; ++nb) {
        acc[0][nb] = wmma_bf16(a0, bf[nb], acc[0][nb]);
        acc[1][nb] = wmma_bf16(a1, bf[nb], acc[1][nb]);
      }
    }
    wait_async_copies();
    __syncthreads();
  }

  // C/D layout: N = lane&15, M = e + 8*(lane>>4)
  #pragma unroll
  for (int mi = 0; mi < 2; ++mi)
    #pragma unroll
    for (int nb = 0; nb < 4; ++nb)
      #pragma unroll
      for (int e = 0; e < 8; ++e) {
        int row = m0 + wm + mi * 16 + e + (lhi << 3);
        int col = n0 + wn + nb * 16 + llo;
        float v = acc[mi][nb][e] + bias[col];
        if (RELU) v = fmaxf(v, 0.f);
        size_t o = (size_t)row * N + col;
        if (HASRES) v += resid[o];
        if (WRITEF) outf[o] = v;
        if (WRITEBF) outbf[o] = f32_bf16(v);
      }
}

// ---------------------------------------------------------------------------
// Flash attention: one wave per (b, h, 16-row q tile); kv tiles of 32.
// All fragment gathers are 16B vector loads. Single-wave workgroup: LDS ops
// from one wave are processed in order, so the P store->load relay needs no
// barrier (compiler inserts s_wait_dscnt for the register dependence).
// ---------------------------------------------------------------------------
__global__ __launch_bounds__(32)
void flash_attn_kernel(const unsigned short* __restrict__ Q,
                       const unsigned short* __restrict__ Kb,
                       const unsigned short* __restrict__ VT,
                       unsigned short* __restrict__ O, int S) {
  const int q0   = blockIdx.x * 16;
  const int h    = blockIdx.y;
  const int bz   = blockIdx.z;
  const int lane = threadIdx.x;
  const int lhi  = lane >> 4, llo = lane & 15;
  const size_t baseRow = (size_t)bz * S;
  const int hc = h * HDIM;
  const float scale = 0.125f;  // 1/sqrt(64)

  __shared__ unsigned short P[16 * 32];  // row stride 64B: b128-aligned

  FragBF qf[2];
  {
    const u32x4* qp = (const u32x4*)(Q + (baseRow + q0 + llo) * D_MODEL + hc);
    #pragma unroll
    for (int f = 0; f < 2; ++f) {
      qf[f].q[0] = qp[f * 4 + 2 * lhi];
      qf[f].q[1] = qp[f * 4 + 2 * lhi + 1];
    }
  }

  v8f oacc[4];
  #pragma unroll
  for (int n = 0; n < 4; ++n) oacc[n] = (v8f){};
  float mrow[8], lrow[8];
  #pragma unroll
  for (int e = 0; e < 8; ++e) { mrow[e] = -1e30f; lrow[e] = 0.f; }

  for (int kv0 = 0; kv0 < S; kv0 += 32) {
    // ---- scores 16x32: two 16x16 C frags, K-dim = HD in two 32-chunks
    v8f s0 = {}, s1 = {};
    const u32x4* kp0 = (const u32x4*)(Kb + (baseRow + kv0 + llo) * D_MODEL + hc);
    const u32x4* kp1 = (const u32x4*)(Kb + (baseRow + kv0 + 16 + llo) * D_MODEL + hc);
    #pragma unroll
    for (int f = 0; f < 2; ++f) {
      FragBF kf0, kf1;
      kf0.q[0] = kp0[f * 4 + 2 * lhi]; kf0.q[1] = kp0[f * 4 + 2 * lhi + 1];
      kf1.q[0] = kp1[f * 4 + 2 * lhi]; kf1.q[1] = kp1[f * 4 + 2 * lhi + 1];
      s0 = wmma_bf16(qf[f], kf0, s0);
      s1 = wmma_bf16(qf[f], kf1, s1);
    }

    // ---- online softmax (row r lives as element e within 16-lane group)
    float alpha[8];
    #pragma unroll
    for (int e = 0; e < 8; ++e) {
      float a = s0[e] * scale, b = s1[e] * scale;
      float mx = fmaxf(a, b);
      #pragma unroll
      for (int o = 8; o > 0; o >>= 1) mx = fmaxf(mx, __shfl_xor(mx, o, 32));
      float mnew = fmaxf(mrow[e], mx);
      float p0 = __expf(a - mnew), p1 = __expf(b - mnew);
      float rsum = p0 + p1;
      #pragma unroll
      for (int o = 8; o > 0; o >>= 1) rsum += __shfl_xor(rsum, o, 32);
      alpha[e] = __expf(mrow[e] - mnew);
      lrow[e]  = lrow[e] * alpha[e] + rsum;
      mrow[e]  = mnew;
      int row = e + (lhi << 3);
      P[row * 32 + llo]      = f32_bf16(p0);
      P[row * 32 + 16 + llo] = f32_bf16(p1);
    }
    #pragma unroll
    for (int n = 0; n < 4; ++n)
      #pragma unroll
      for (int e = 0; e < 8; ++e) oacc[n][e] *= alpha[e];

    // ---- P (A frag, ds_load_b128 x2) @ V (B frags from kv-major vT)
    FragBF pf;
    {
      const u32x4* pp = (const u32x4*)(P + llo * 32);
      pf.q[0] = pp[2 * lhi];
      pf.q[1] = pp[2 * lhi + 1];
    }
    #pragma unroll
    for (int n = 0; n < 4; ++n) {
      FragBF vf;
      const u32x4* vp = (const u32x4*)(VT + ((size_t)bz * D_MODEL + hc + n * 16 + llo) * S
                                          + kv0 + 16 * lhi);
      vf.q[0] = vp[0];
      vf.q[1] = vp[1];
      oacc[n] = wmma_bf16(pf, vf, oacc[n]);
    }
  }

  // ---- normalize + write bf16 (feeds Wo projection GEMM)
  #pragma unroll
  for (int n = 0; n < 4; ++n)
    #pragma unroll
    for (int e = 0; e < 8; ++e) {
      int row = q0 + e + (lhi << 3);
      int col = hc + n * 16 + llo;
      O[(baseRow + row) * D_MODEL + col] = f32_bf16(oacc[n][e] / lrow[e]);
    }
}

// ---------------------------------------------------------------------------
// Host orchestration. Workspace budget ~192 MB (bf16 weights + activations).
// ---------------------------------------------------------------------------
extern "C" void kernel_launch(void* const* d_in, const int* in_sizes, int n_in,
                              void* d_out, int out_size, void* d_ws, size_t ws_size,
                              hipStream_t stream) {
  const float* x    = (const float*)d_in[0];
  const float* Wq   = (const float*)d_in[1];
  const float* bq   = (const float*)d_in[2];
  const float* Wk   = (const float*)d_in[3];
  const float* bk   = (const float*)d_in[4];
  const float* Wv   = (const float*)d_in[5];
  const float* bv   = (const float*)d_in[6];
  const float* Wo   = (const float*)d_in[7];
  const float* bo   = (const float*)d_in[8];
  const float* W1   = (const float*)d_in[9];
  const float* b1   = (const float*)d_in[10];
  const float* W2   = (const float*)d_in[11];
  const float* b2   = (const float*)d_in[12];
  const float* ln1g = (const float*)d_in[13];
  const float* ln1b = (const float*)d_in[14];
  const float* ln2g = (const float*)d_in[15];
  const float* ln2b = (const float*)d_in[16];

  const int Bb = 2, S = 2048;
  const int Mrows = Bb * S;              // 4096 token rows
  const int Dm = D_MODEL, Dff = D_FF;

  char* ws = (char*)d_ws;
  size_t off = 0;
  auto carve = [&](size_t bytes) {
    char* p = ws + off;
    off += (bytes + 255) & ~(size_t)255;
    return p;
  };
  // K-major (transposed) bf16 weights
  unsigned short* wqT = (unsigned short*)carve((size_t)NLAYERS * Dm * Dm  * 2);
  unsigned short* wkT = (unsigned short*)carve((size_t)NLAYERS * Dm * Dm  * 2);
  unsigned short* wvT = (unsigned short*)carve((size_t)NLAYERS * Dm * Dm  * 2);
  unsigned short* woT = (unsigned short*)carve((size_t)NLAYERS * Dm * Dm  * 2);
  unsigned short* w1T = (unsigned short*)carve((size_t)NLAYERS * Dm * Dff * 2);
  unsigned short* w2T = (unsigned short*)carve((size_t)NLAYERS * Dff * Dm * 2);
  float*          xbuf  = (float*)         carve((size_t)Mrows * Dm * 4);
  unsigned short* hbf   = (unsigned short*)carve((size_t)Mrows * Dm * 2);
  unsigned short* qbf   = (unsigned short*)carve((size_t)Mrows * Dm * 2);
  unsigned short* kbf   = (unsigned short*)carve((size_t)Mrows * Dm * 2);
  unsigned short* vbf   = (unsigned short*)carve((size_t)Mrows * Dm * 2);
  unsigned short* vtbf  = (unsigned short*)carve((size_t)Mrows * Dm * 2);
  unsigned short* obf   = (unsigned short*)carve((size_t)Mrows * Dm * 2);
  unsigned short* fbf   = (unsigned short*)carve((size_t)Mrows * Dff * 2);

  // convert + transpose weights: src [L][K][N] -> dst [L][N][K]
  auto cvtT = [&](const float* src, unsigned short* dst, int K, int N) {
    cvt_transpose_bf16_kernel<<<dim3(N / 32, K / 32, NLAYERS), dim3(32, 8), 0, stream>>>(
        src, dst, K, N);
  };
  cvtT(Wq, wqT, Dm, Dm);
  cvtT(Wk, wkT, Dm, Dm);
  cvtT(Wv, wvT, Dm, Dm);
  cvtT(Wo, woT, Dm, Dm);
  cvtT(W1, w1T, Dm, Dff);
  cvtT(W2, w2T, Dff, Dm);

  // GEMM variants: <RELU, HASRES, WRITEF, WRITEBF>
  auto gemm_bf = [&](const unsigned short* A, const unsigned short* Wt, const float* bias,
                     unsigned short* outbf, int N, int K) {
    dim3 grid(N / 128, Mrows / 128);
    gemm_bf16_kernel<false, false, false, true>
        <<<grid, 256, 0, stream>>>(A, Wt, bias, nullptr, nullptr, outbf, N, K);
  };
  auto gemm_relu_bf = [&](const unsigned short* A, const unsigned short* Wt, const float* bias,
                          unsigned short* outbf, int N, int K) {
    dim3 grid(N / 128, Mrows / 128);
    gemm_bf16_kernel<true, false, false, true>
        <<<grid, 256, 0, stream>>>(A, Wt, bias, nullptr, nullptr, outbf, N, K);
  };
  auto gemm_res_f32 = [&](const unsigned short* A, const unsigned short* Wt, const float* bias,
                          const float* resid, float* outf, int N, int K) {
    dim3 grid(N / 128, Mrows / 128);
    gemm_bf16_kernel<false, true, true, false>
        <<<grid, 256, 0, stream>>>(A, Wt, bias, resid, outf, nullptr, N, K);
  };

  const float* xin = x;  // layer-0 residual input (never mutated)
  for (int i = 0; i < NLAYERS; ++i) {
    // --- attention sublayer ---
    layernorm_bf16_kernel<<<Mrows, 256, 0, stream>>>(xin, ln1g + i * Dm, ln1b + i * Dm, hbf);
    gemm_bf(hbf, wqT + (size_t)i * Dm * Dm, bq + i * Dm, qbf, Dm, Dm);
    gemm_bf(hbf, wkT + (size_t)i * Dm * Dm, bk + i * Dm, kbf, Dm, Dm);
    gemm_bf(hbf, wvT + (size_t)i * Dm * Dm, bv + i * Dm, vbf, Dm, Dm);
    transpose_bf16_kernel<<<dim3(Dm / 32, S / 32, Bb), dim3(32, 8), 0, stream>>>(
        vbf, vtbf, S, Dm);
    flash_attn_kernel<<<dim3(S / 16, NHEADS, Bb), 32, 0, stream>>>(qbf, kbf, vtbf, obf, S);
    gemm_res_f32(obf, woT + (size_t)i * Dm * Dm, bo + i * Dm, xin, xbuf, Dm, Dm);

    // --- feedforward sublayer ---
    layernorm_bf16_kernel<<<Mrows, 256, 0, stream>>>(xbuf, ln2g + i * Dm, ln2b + i * Dm, hbf);
    gemm_relu_bf(hbf, w1T + (size_t)i * Dm * Dff, b1 + i * Dff, fbf, Dff, Dm);
    float* xdst = (i == NLAYERS - 1) ? (float*)d_out : xbuf;
    gemm_res_f32(fbf, w2T + (size_t)i * Dff * Dm, b2 + i * Dm, xbuf, xdst, Dm, Dff);

    xin = xbuf;
  }
  (void)in_sizes; (void)n_in; (void)out_size; (void)ws_size;
}
